// HCA_21526376087644
// MI455X (gfx1250) — compile-verified, matrix operand order checked by reading, not run on hardware
//
#include <hip/hip_runtime.h>

// ---------------------------------------------------------------------------
// GeneralGNN forward for MI455X (gfx1250, wave32, WMMA + async LDS staging).
//   xb  = bf16(x)                          one-pass convert
//   h   = prelu(xb @ W_pre + b_pre)        GEMM1  (v_wmma_f32_16x16x32_bf16)
//   m   = hb @ W_mp + b_mp                 GEMM2
//   agg = scatter-add m[senders] -> rcv    f32 atomics (m/agg L2-resident, 51MB)
//   h2  = prelu(agg) + h                   fused elementwise -> bf16
//   out = relu(h2b @ W_post + b_post)      GEMM3
// GEMM: B-fragments double-buffered into LDS via GLOBAL_LOAD_ASYNC_TO_LDS_B128
// (ASYNCcnt + s_wait_asynccnt), consumed with ds_load_b128; A streamed from
// global. f32 accumulate/bias/PReLU/skip/aggregation, bf16 matmul inputs.
// ---------------------------------------------------------------------------

typedef __attribute__((ext_vector_type(16))) __bf16         v16bf;
typedef __attribute__((ext_vector_type(8)))  float          v8f;
typedef __attribute__((ext_vector_type(16))) unsigned short v16u;
typedef __attribute__((ext_vector_type(8)))  unsigned short v8u;
typedef __attribute__((ext_vector_type(4)))  unsigned short v4u;
typedef __attribute__((ext_vector_type(4)))  int            v4i;

// AS-qualified pointer typedefs (attribute must live in a typedef to apply)
typedef __attribute__((address_space(1))) v4i* gp_b128;   // global int4*
typedef __attribute__((address_space(3))) v4i* lp_b128;   // LDS int4*

#define DIM 256   // F == H == 256

__device__ __forceinline__ unsigned short f2bf(float f) {
  unsigned u = __builtin_bit_cast(unsigned, f);
  u += 0x7FFFu + ((u >> 16) & 1u);          // round-to-nearest-even
  return (unsigned short)(u >> 16);
}

// one 16-byte async global->LDS chunk (ASYNCcnt-tracked DMA, no VGPR data)
__device__ __forceinline__ void async_b128(const unsigned short* g, unsigned short* l) {
  __builtin_amdgcn_global_load_async_to_lds_b128(
      (gp_b128)(v4i*)(unsigned short*)g, (lp_b128)(v4i*)l, 0, 0);
}

// ---------------------------------------------------------------------------
// f32 -> bf16 bulk convert (vectorized: b128 in, b64 out)
// ---------------------------------------------------------------------------
__global__ __launch_bounds__(256) void cvt_kernel(const float4* __restrict__ in,
                                                  v4u* __restrict__ outv, int n4) {
  int i = blockIdx.x * blockDim.x + threadIdx.x;
  if (i >= n4) return;
  const float4 f = in[i];
  v4u o;
  o[0] = f2bf(f.x); o[1] = f2bf(f.y); o[2] = f2bf(f.z); o[3] = f2bf(f.w);
  outv[i] = o;
}

// ---------------------------------------------------------------------------
// Prepack 3 weight matrices [256(K) x 256(N), row-major f32] into bf16 WMMA
// B fragments. Fragment (kt, nt) covers K=kt*32..+31, N=nt*16..+15, stored
// lane-major: 32 lanes x 16 bf16, matching the 16-bit 16x32 operand layout
// (lane half selects K chunk {0-7,16-23} vs {8-15,24-31}). One K-stage (kt)
// is 16 contiguous fragments = 16KB, ready for linear async copy to LDS.
// ---------------------------------------------------------------------------
__global__ __launch_bounds__(256) void prepack_kernel(
    const float* __restrict__ W0, const float* __restrict__ W1,
    const float* __restrict__ W2, unsigned short* __restrict__ Wp) {
  int idx = blockIdx.x * blockDim.x + threadIdx.x;
  if (idx >= 3 * 65536) return;
  int w    = idx >> 16;
  int rem  = idx & 65535;
  int frag = rem >> 9;          // 0..127  (kt*16 + nt)
  int lane = (rem >> 4) & 31;
  int e    = rem & 15;
  int kt   = frag >> 4;
  int nt   = frag & 15;
  int col  = nt * 16 + (lane & 15);
  int k    = kt * 32 + ((lane >> 4) * 8) + (e < 8 ? e : e + 8);
  const float* W = (w == 0) ? W0 : (w == 1) ? W1 : W2;
  Wp[idx] = f2bf(W[k * DIM + col]);
}

// ---------------------------------------------------------------------------
// GEMM: C[M,256] = act(A_bf16[M,256] @ Wpack + bias).
// Block = 256 threads = 8 waves -> 64x256 output tile.
// Wave(wrow 0..3, wcol 0..1) -> 16 rows x 128 cols = 8 WMMA accumulators.
// B tiles (16KB per K-stage) double-buffered in LDS via async loads.
// ACT: 0 none, 1 prelu(alpha), 2 relu.  WRITE_BF16: also emit bf16 copy.
// ---------------------------------------------------------------------------
template <int ACT, int WRITE_BF16>
__global__ __launch_bounds__(256) void gemm_kernel(
    const unsigned short* __restrict__ Ab, const unsigned short* __restrict__ Wp,
    const float* __restrict__ bias, const float* __restrict__ alpha,
    float* __restrict__ Cf, unsigned short* __restrict__ Cb, int M) {
  __shared__ unsigned short Bsm[2][8192];   // 2 x 16KB B-stage double buffer

  const int tid   = threadIdx.x;
  const int lane  = tid & 31;
  const int wid   = tid >> 5;
  const int wrow  = wid >> 1;
  const int wcolt = (wid & 1) * 8;      // column-tile base (0 or 8)
  const int mloc  = lane & 15;
  const int lhalf = lane >> 4;
  const int brow  = blockIdx.x * 64;

  int rowA = brow + wrow * 16 + mloc;
  if (rowA >= M) rowA = M - 1;          // clamp: keep EXEC all-ones for WMMA

  // Issue async copy of stage 0 (16KB spread across the block: 4 x b128/thread)
#pragma unroll
  for (int c = 0; c < 4; ++c) {
    const int eo = (c * 256 + tid) * 8;   // ushort offset of a 16B chunk
    async_b128(Wp + eo, &Bsm[0][eo]);
  }

  v8f acc[8];
#pragma unroll
  for (int i = 0; i < 8; ++i) acc[i] = (v8f){0.f, 0.f, 0.f, 0.f, 0.f, 0.f, 0.f, 0.f};

#pragma unroll
  for (int kt = 0; kt < 8; ++kt) {
    if (kt < 7) {
      // stream next K-stage into the other LDS buffer while we compute
      const unsigned short* g = Wp + (size_t)(kt + 1) * 8192;
#pragma unroll
      for (int c = 0; c < 4; ++c) {
        const int eo = (c * 256 + tid) * 8;
        async_b128(g + eo, &Bsm[(kt + 1) & 1][eo]);
      }
      // async loads complete in order: <=4 outstanding => stage kt landed
      asm volatile("s_wait_asynccnt 0x4" ::: "memory");
    } else {
      asm volatile("s_wait_asynccnt 0x0" ::: "memory");
    }
    __syncthreads();

    // A fragment: two 16B row-major chunks per lane
    const int kbase = kt * 32 + lhalf * 8;
    const unsigned short* Apt = Ab + (size_t)rowA * DIM + kbase;
    const v8u a0 = *(const v8u*)Apt;          // K chunk {0..7}  (rel. lane half)
    const v8u a1 = *(const v8u*)(Apt + 16);   // K chunk {16..23}
    v16u au;
#pragma unroll
    for (int i = 0; i < 8; ++i) { au[i] = a0[i]; au[8 + i] = a1[i]; }
    const v16bf a = __builtin_bit_cast(v16bf, au);

#pragma unroll
    for (int nt = 0; nt < 8; ++nt) {
      const unsigned short* bp = &Bsm[kt & 1][(wcolt + nt) * 512 + lane * 16];
      const v16u bu = *(const v16u*)bp;       // ds_load_b128 x2
      const v16bf b = __builtin_bit_cast(v16bf, bu);
      acc[nt] = __builtin_amdgcn_wmma_f32_16x16x32_bf16(
          false, a, false, b, (short)0, acc[nt], false, false);
    }
    __syncthreads();   // all waves done reading before this buffer is refilled
  }

  // Epilogue: C layout per ISA — lane half selects M=j vs M=j+8, N = lane&15.
#pragma unroll
  for (int nt = 0; nt < 8; ++nt) {
    const int col  = (wcolt + nt) * 16 + mloc;
    const float bc = bias[col];
    const float ac = (ACT == 1) ? alpha[col] : 0.f;
#pragma unroll
    for (int j = 0; j < 8; ++j) {
      const int r = brow + wrow * 16 + lhalf * 8 + j;
      if (r < M) {
        float v = acc[nt][j] + bc;
        if (ACT == 1) v = (v > 0.f) ? v : ac * v;
        if (ACT == 2) v = (v > 0.f) ? v : 0.f;
        Cf[(size_t)r * DIM + col] = v;
        if (WRITE_BF16) Cb[(size_t)r * DIM + col] = f2bf(v);
      }
    }
  }
}

// ---------------------------------------------------------------------------
// agg = 0
// ---------------------------------------------------------------------------
__global__ __launch_bounds__(256) void zero_kernel(float4* __restrict__ p, int n4) {
  int i = blockIdx.x * blockDim.x + threadIdx.x;
  if (i < n4) p[i] = make_float4(0.f, 0.f, 0.f, 0.f);
}

// ---------------------------------------------------------------------------
// Edge scatter: one wave per edge; lanes cover 256 features as float4 pairs.
// agg (51MB) and m (51MB) both fit in 192MB L2 -> atomics resolve at L2.
// ---------------------------------------------------------------------------
__global__ __launch_bounds__(256) void scatter_kernel(
    const float* __restrict__ m, const int* __restrict__ snd,
    const int* __restrict__ rcv, float* __restrict__ agg, int E) {
  const int lane = threadIdx.x & 31;
  const int w    = blockIdx.x * (blockDim.x >> 5) + (threadIdx.x >> 5);
  const int nw   = gridDim.x * (blockDim.x >> 5);
  for (int e = w; e < E; e += nw) {
    const int en = e + nw;
    if (en < E) {  // prefetch next iteration's source row (global_prefetch_b8)
      __builtin_prefetch(m + (size_t)snd[en] * DIM, 0, 1);
    }
    const int s = snd[e];
    const int r = rcv[e];
    const float4* src = (const float4*)(m + (size_t)s * DIM);
    float* dst = agg + (size_t)r * DIM;
#pragma unroll
    for (int it = 0; it < 2; ++it) {
      const int c4 = lane + it * 32;
      const float4 v = src[c4];
      __hip_atomic_fetch_add(dst + c4 * 4 + 0, v.x, __ATOMIC_RELAXED, __HIP_MEMORY_SCOPE_AGENT);
      __hip_atomic_fetch_add(dst + c4 * 4 + 1, v.y, __ATOMIC_RELAXED, __HIP_MEMORY_SCOPE_AGENT);
      __hip_atomic_fetch_add(dst + c4 * 4 + 2, v.z, __ATOMIC_RELAXED, __HIP_MEMORY_SCOPE_AGENT);
      __hip_atomic_fetch_add(dst + c4 * 4 + 3, v.w, __ATOMIC_RELAXED, __HIP_MEMORY_SCOPE_AGENT);
    }
  }
}

// ---------------------------------------------------------------------------
// h2 = prelu(agg, alpha_mp) + h   (f32 math, bf16 output for GEMM3)
// ---------------------------------------------------------------------------
__global__ __launch_bounds__(256) void fuse_kernel(
    const float4* __restrict__ agg, const float4* __restrict__ h,
    const float* __restrict__ alpha, unsigned short* __restrict__ h2b, int n4) {
  int i = blockIdx.x * blockDim.x + threadIdx.x;
  if (i >= n4) return;
  const float4 a  = agg[i];
  const float4 hh = h[i];
  const int colb  = (i * 4) & (DIM - 1);
  const float4 al = *(const float4*)(alpha + colb);
  const float v0 = ((a.x > 0.f) ? a.x : al.x * a.x) + hh.x;
  const float v1 = ((a.y > 0.f) ? a.y : al.y * a.y) + hh.y;
  const float v2 = ((a.z > 0.f) ? a.z : al.z * a.z) + hh.z;
  const float v3 = ((a.w > 0.f) ? a.w : al.w * a.w) + hh.w;
  v4u o;
  o[0] = f2bf(v0); o[1] = f2bf(v1); o[2] = f2bf(v2); o[3] = f2bf(v3);
  *(v4u*)(h2b + (size_t)i * 4) = o;
}

// ---------------------------------------------------------------------------
extern "C" void kernel_launch(void* const* d_in, const int* in_sizes, int n_in,
                              void* d_out, int out_size, void* d_ws, size_t ws_size,
                              hipStream_t stream) {
  const float* x      = (const float*)d_in[0];
  const int*   snd    = (const int*)d_in[1];
  const int*   rcv    = (const int*)d_in[2];
  const float* W_pre  = (const float*)d_in[3];
  const float* b_pre  = (const float*)d_in[4];
  const float* a_pre  = (const float*)d_in[5];
  const float* W_mp   = (const float*)d_in[6];
  const float* b_mp   = (const float*)d_in[7];
  const float* a_mp   = (const float*)d_in[8];
  const float* W_post = (const float*)d_in[9];
  const float* b_post = (const float*)d_in[10];
  float* out = (float*)d_out;

  const int N = in_sizes[0] / DIM;   // 50000
  const int E = in_sizes[1];         // 800000
  const size_t nf = (size_t)N * DIM;

  // Workspace carve-up (~180 MB): Wpack | h_f32 | h_bf16(->h2_bf16) | m | agg
  // xb (bf16 x) aliases m's region: xb is dead before GEMM2 writes m.
  char* ws = (char*)d_ws;
  size_t off = 0;
  unsigned short* Wp = (unsigned short*)(ws + off); off += 524288;       // 3*128KB packed + pad
  float*          h  = (float*)(ws + off);          off += nf * 4;
  unsigned short* hb = (unsigned short*)(ws + off); off += nf * 2;       // reused as h2 bf16
  float*          m  = (float*)(ws + off);          off += nf * 4;
  float*          agg= (float*)(ws + off);          off += nf * 4;
  unsigned short* xb = (unsigned short*)m;          // alias (see above)

  const int mblocks = (N + 63) / 64;
  const int n4      = (int)(nf / 4);

  prepack_kernel<<<(3 * 65536 + 255) / 256, 256, 0, stream>>>(W_pre, W_mp, W_post, Wp);
  cvt_kernel<<<(n4 + 255) / 256, 256, 0, stream>>>((const float4*)x, (v4u*)xb, n4);

  // GEMM1: h = prelu(x @ W_pre + b_pre); emit f32 (skip path) + bf16 (next GEMM)
  gemm_kernel<1, 1><<<mblocks, 256, 0, stream>>>(xb, Wp, b_pre, a_pre, h, hb, N);

  // GEMM2: m = h @ W_mp + b_mp   (overwrites xb alias -- xb no longer needed)
  gemm_kernel<0, 0><<<mblocks, 256, 0, stream>>>(hb, Wp + 65536, b_mp, nullptr, m, nullptr, N);

  // agg = segment_sum(m[senders], receivers)
  zero_kernel<<<(n4 + 255) / 256, 256, 0, stream>>>((float4*)agg, n4);
  scatter_kernel<<<12500, 256, 0, stream>>>(m, snd, rcv, agg, E);

  // h2 = prelu(agg) + h  -> bf16 (overwrites hb)
  fuse_kernel<<<(n4 + 255) / 256, 256, 0, stream>>>((const float4*)agg, (const float4*)h,
                                                    a_mp, hb, n4);

  // GEMM3: out = relu(h2 @ W_post + b_post)
  gemm_kernel<2, 0><<<mblocks, 256, 0, stream>>>(hb, Wp + 2 * 65536, b_post, nullptr, out, nullptr, N);
}
